// ConditionalVoronoiFlowDistribution_10196252361180
// MI455X (gfx1250) — matrix operand order repaired and checked by reading, not compiled, not run on hardware
//
#include <hip/hip_runtime.h>
#include <hip/hip_bf16.h>

// ---------------------------------------------------------------------------
// Problem constants
// ---------------------------------------------------------------------------
#define BQ    8192          // batch
#define DQ    32            // variables
#define KQ    16            // dims per variable
#define CQ    64            // cond embedding dim
#define NCQ   256           // codebook size
#define NBQ   4             // flow blocks
#define HQ    1024          // coupling hidden
#define FDQ   512           // flow dim = D*K
#define CINQ  2560          // coupling input = D*(K+C)
#define CCQ   2048          // cond-affine input = D*C

typedef __attribute__((ext_vector_type(16))) __bf16 v16bf;
typedef __attribute__((ext_vector_type(8)))  float  v8f;

union Frag { uint4 u[2]; v16bf v; };

// ---------------------------------------------------------------------------
// CDNA5 async global->LDS path (guarded; falls back to reg-staged pipeline)
// Builtin signature (from toolchain diagnostic): params are pointers to
// int __attribute__((vector_size(16))) in AS(1) (src) / AS(3) (dst).
// ---------------------------------------------------------------------------
#if defined(__has_builtin)
#if __has_builtin(__builtin_amdgcn_global_load_async_to_lds_b128)
#define ASYNC_CP 1
#endif
#endif

#ifdef ASYNC_CP
typedef int v4i_t __attribute__((vector_size(16)));
typedef __attribute__((address_space(1))) v4i_t gv4i;
typedef __attribute__((address_space(3))) v4i_t lv4i;
__device__ __forceinline__ void async_cp16(const void* g, void* l) {
    __builtin_amdgcn_global_load_async_to_lds_b128(
        (gv4i*)(unsigned long long)g,
        (lv4i*)(unsigned int)(unsigned long long)l, 0, 0);
}
#if __has_builtin(__builtin_amdgcn_s_wait_asynccnt)
#define WAIT_ASYNC() __builtin_amdgcn_s_wait_asynccnt(0)
#else
#define WAIT_ASYNC() asm volatile("s_wait_asynccnt 0x0" ::: "memory")
#endif
#endif

// ---------------------------------------------------------------------------
// Helpers
// ---------------------------------------------------------------------------
__device__ __forceinline__ unsigned short f2bf(float x) {
    unsigned int u = __float_as_uint(x);
    u = (u + 0x7FFFu + ((u >> 16) & 1u)) >> 16;
    return (unsigned short)u;
}

__device__ __forceinline__ float fast_tanh(float x) {
#if defined(__has_builtin)
#if __has_builtin(__builtin_amdgcn_tanhf)
    return __builtin_amdgcn_tanhf(x);
#else
    return tanhf(x);
#endif
#else
    return tanhf(x);
#endif
}

__device__ __forceinline__ float geluf(float x) {
    // jax.nn.gelu default (approximate=True, tanh form)
    float x3 = x * x * x;
    return 0.5f * x * (1.0f + fast_tanh(0.7978845608028654f * (x + 0.044715f * x3)));
}

__device__ __forceinline__ float swishf(float x) {
    // x * sigmoid(x * softplus(0.5)); softplus(0.5) = log1p(exp(0.5))
    const float sp = 0.9740769841801067f;
    return x * (1.0f / (1.0f + __expf(-x * sp)));
}

__device__ __forceinline__ bool mask_keep(int f, int mt) {
    if (mt == 0) return (f & 1) == 0;
    if (mt == 1) return (f & 1) == 1;
    if (mt == 2) return f < (FDQ / 2);
    return f >= (FDQ / 2);
}

// ---------------------------------------------------------------------------
// f32 [K,N] -> bf16 transposed [N,K] (tiled via LDS)
// ---------------------------------------------------------------------------
__global__ __launch_bounds__(256)
void cvt_transpose_kernel(const float* __restrict__ src, unsigned short* __restrict__ dst,
                          int K, int N) {
    __shared__ float tile[32][33];
    const int tx = threadIdx.x & 31;
    const int ty = threadIdx.x >> 5;
    const int kbase = blockIdx.y * 32;
    const int nbase = blockIdx.x * 32;
#pragma unroll
    for (int i = 0; i < 4; ++i)
        tile[ty + 8 * i][tx] = src[(size_t)(kbase + ty + 8 * i) * N + nbase + tx];
    __syncthreads();
#pragma unroll
    for (int i = 0; i < 4; ++i)
        dst[(size_t)(nbase + ty + 8 * i) * K + kbase + tx] = f2bf(tile[tx][ty + 8 * i]);
}

// ---------------------------------------------------------------------------
// Gather cond = emb[tokens] into bf16 cond buffer and the cond-slots of zin
// ---------------------------------------------------------------------------
__global__ __launch_bounds__(256) void gather_cond_kernel(const int* __restrict__ tokens,
                                                          const float* __restrict__ emb,
                                                          unsigned short* __restrict__ cond,
                                                          unsigned short* __restrict__ zin) {
    int i = blockIdx.x * 256 + threadIdx.x;   // over B*D*C
    int b = i >> 11;
    int rem = i & 2047;
    int d = rem >> 6;
    int c = rem & 63;
    int tok = tokens[b * DQ + d];
    unsigned short h = f2bf(emb[(size_t)tok * CQ + c]);
    cond[(size_t)b * CCQ + d * CQ + c] = h;
    zin[(size_t)b * CINQ + d * (KQ + CQ) + KQ + c] = h;
}

// ---------------------------------------------------------------------------
// Write masked-z strips of zin (bf16) for one block
// ---------------------------------------------------------------------------
__global__ __launch_bounds__(256) void zin_zm_kernel(const float* __restrict__ z,
                                                     unsigned short* __restrict__ zin,
                                                     int mt) {
    int i = blockIdx.x * 256 + threadIdx.x;   // over B*FD
    int b = i >> 9;
    int f = i & 511;
    int d = f >> 4;
    int k = f & 15;
    float v = mask_keep(f, mt) ? z[i] : 0.0f;
    zin[(size_t)b * CINQ + d * (KQ + CQ) + k] = f2bf(v);
}

// ---------------------------------------------------------------------------
// WMMA GEMM (bf16 x bf16 -> f32), B pre-transposed [N,K], fused activation.
// out[M,N] = act(A[M,K] @ W^T + bias); tile 128x64, K step 32, double-buffered.
// ACT: 0 = GELU, 1 = Swish
// ---------------------------------------------------------------------------
template <int ACT>
__global__ __launch_bounds__(256)
void gemm_act_kernel(const unsigned short* __restrict__ A,    // [M,K]
                     const unsigned short* __restrict__ Wt,   // [N,K]
                     const float* __restrict__ bias,
                     unsigned short* __restrict__ out,
                     int K, int N) {
    __shared__ __align__(16) unsigned short As[2][128 * 48];
    __shared__ __align__(16) unsigned short Bt[2][64 * 48];

    const int tid  = threadIdx.x;
    const int lane = tid & 31;
    const int wave = tid >> 5;
    const int wr = wave >> 1, wc = wave & 1;
    const int row0 = blockIdx.y * 128;
    const int col0 = blockIdx.x * 64;
    const int m    = lane & 15;
    const int half = lane >> 4;

    const int srow = tid >> 2;           // 0..63
    const int scol = (tid & 3) * 8;      // 0,8,16,24
    const unsigned short* gA0 = A  + (size_t)(row0 + srow) * K + scol;
    const unsigned short* gA1 = A  + (size_t)(row0 + 64 + srow) * K + scol;
    const unsigned short* gB0 = Wt + (size_t)(col0 + srow) * K + scol;
    const int lA0 = srow * 48 + scol;
    const int lA1 = (64 + srow) * 48 + scol;
    const int lB0 = srow * 48 + scol;

    v8f acc[2][2] = {};
    const int nK = K >> 5;
    int buf = 0;

#ifdef ASYNC_CP
    async_cp16(gA0, &As[0][lA0]);
    async_cp16(gA1, &As[0][lA1]);
    async_cp16(gB0, &Bt[0][lB0]);
#else
    uint4 rA0 = *reinterpret_cast<const uint4*>(gA0);
    uint4 rA1 = *reinterpret_cast<const uint4*>(gA1);
    uint4 rB0 = *reinterpret_cast<const uint4*>(gB0);
    *reinterpret_cast<uint4*>(&As[0][lA0]) = rA0;
    *reinterpret_cast<uint4*>(&As[0][lA1]) = rA1;
    *reinterpret_cast<uint4*>(&Bt[0][lB0]) = rB0;
#endif

    for (int kt = 0; kt < nK; ++kt) {
        const bool more = (kt + 1) < nK;
        const int koff = (kt + 1) * 32;
#ifdef ASYNC_CP
        WAIT_ASYNC();                 // tile kt landed in buf
        __syncthreads();              // visible to all; all done reading buf^1
        if (more) {                   // DMA tile kt+1 while computing tile kt
            async_cp16(gA0 + koff, &As[buf ^ 1][lA0]);
            async_cp16(gA1 + koff, &As[buf ^ 1][lA1]);
            async_cp16(gB0 + koff, &Bt[buf ^ 1][lB0]);
        }
#else
        __syncthreads();
        if (more) {                   // issue global loads early, store after compute
            rA0 = *reinterpret_cast<const uint4*>(gA0 + koff);
            rA1 = *reinterpret_cast<const uint4*>(gA1 + koff);
            rB0 = *reinterpret_cast<const uint4*>(gB0 + koff);
        }
#endif
        Frag a[2], bfr[2];
#pragma unroll
        for (int i = 0; i < 2; ++i) {
            const unsigned short* p = &As[buf][(wr * 32 + i * 16 + m) * 48 + half * 8];
            a[i].u[0] = *reinterpret_cast<const uint4*>(p);
            a[i].u[1] = *reinterpret_cast<const uint4*>(p + 16);
            const unsigned short* q = &Bt[buf][(wc * 32 + i * 16 + m) * 48 + half * 8];
            bfr[i].u[0] = *reinterpret_cast<const uint4*>(q);
            bfr[i].u[1] = *reinterpret_cast<const uint4*>(q + 16);
        }
#pragma unroll
        for (int mi = 0; mi < 2; ++mi)
#pragma unroll
            for (int ni = 0; ni < 2; ++ni)
                acc[mi][ni] = __builtin_amdgcn_wmma_f32_16x16x32_bf16(
                    false, a[mi].v, false, bfr[ni].v, (short)0, acc[mi][ni], false, false);
#ifndef ASYNC_CP
        if (more) {
            *reinterpret_cast<uint4*>(&As[buf ^ 1][lA0]) = rA0;
            *reinterpret_cast<uint4*>(&As[buf ^ 1][lA1]) = rA1;
            *reinterpret_cast<uint4*>(&Bt[buf ^ 1][lB0]) = rB0;
        }
#endif
        buf ^= 1;
    }

#pragma unroll
    for (int mi = 0; mi < 2; ++mi) {
        int rbase = row0 + wr * 32 + mi * 16 + half * 8;
#pragma unroll
        for (int ni = 0; ni < 2; ++ni) {
            int col = col0 + wc * 32 + ni * 16 + m;
            float bv = bias[col];
#pragma unroll
            for (int r = 0; r < 8; ++r) {
                float x = acc[mi][ni][r] + bv;
                x = (ACT == 0) ? geluf(x) : swishf(x);
                out[(size_t)(rbase + r) * N + col] = f2bf(x);
            }
        }
    }
}

// ---------------------------------------------------------------------------
// Dual-half WMMA GEMM for the 3rd nets: computes output cols [c] and [c+512]
// of W^T [1024,K] simultaneously, fuses the flow update, writes z in place.
// EPI 0: ActNorm + conditional affine   EPI 1: masked coupling update
// ---------------------------------------------------------------------------
template <int EPI>
__global__ __launch_bounds__(256)
void gemm_dual_kernel(const unsigned short* __restrict__ A,    // [M,K]
                      const unsigned short* __restrict__ Wt,   // [1024,K]
                      const float* __restrict__ b3,            // [1024]
                      float* __restrict__ zio,                 // [8192,512]
                      const float* __restrict__ alp,
                      const float* __restrict__ abp,
                      int K, int mt_mask) {
    __shared__ __align__(16) unsigned short As[2][128 * 48];
    __shared__ __align__(16) unsigned short Bt[2][128 * 48];   // rows 0..63 half0, 64..127 half1

    const int tid  = threadIdx.x;
    const int lane = tid & 31;
    const int wave = tid >> 5;
    const int wr = wave >> 1, wc = wave & 1;
    const int row0 = blockIdx.y * 128;
    const int col0 = blockIdx.x * 64;
    const int m    = lane & 15;
    const int half = lane >> 4;

    const int srow = tid >> 2;
    const int scol = (tid & 3) * 8;
    const unsigned short* gA0 = A  + (size_t)(row0 + srow) * K + scol;
    const unsigned short* gA1 = A  + (size_t)(row0 + 64 + srow) * K + scol;
    const unsigned short* gB0 = Wt + (size_t)(col0 + srow) * K + scol;
    const unsigned short* gB1 = Wt + (size_t)(col0 + 512 + srow) * K + scol;
    const int lA0 = srow * 48 + scol;
    const int lA1 = (64 + srow) * 48 + scol;

    v8f accS[2][2] = {};
    v8f accT[2][2] = {};
    const int nK = K >> 5;
    int buf = 0;

#ifdef ASYNC_CP
    async_cp16(gA0, &As[0][lA0]);
    async_cp16(gA1, &As[0][lA1]);
    async_cp16(gB0, &Bt[0][lA0]);
    async_cp16(gB1, &Bt[0][lA1]);
#else
    uint4 rA0 = *reinterpret_cast<const uint4*>(gA0);
    uint4 rA1 = *reinterpret_cast<const uint4*>(gA1);
    uint4 rB0 = *reinterpret_cast<const uint4*>(gB0);
    uint4 rB1 = *reinterpret_cast<const uint4*>(gB1);
    *reinterpret_cast<uint4*>(&As[0][lA0]) = rA0;
    *reinterpret_cast<uint4*>(&As[0][lA1]) = rA1;
    *reinterpret_cast<uint4*>(&Bt[0][lA0]) = rB0;
    *reinterpret_cast<uint4*>(&Bt[0][lA1]) = rB1;
#endif

    for (int kt = 0; kt < nK; ++kt) {
        const bool more = (kt + 1) < nK;
        const int koff = (kt + 1) * 32;
#ifdef ASYNC_CP
        WAIT_ASYNC();
        __syncthreads();
        if (more) {
            async_cp16(gA0 + koff, &As[buf ^ 1][lA0]);
            async_cp16(gA1 + koff, &As[buf ^ 1][lA1]);
            async_cp16(gB0 + koff, &Bt[buf ^ 1][lA0]);
            async_cp16(gB1 + koff, &Bt[buf ^ 1][lA1]);
        }
#else
        __syncthreads();
        if (more) {
            rA0 = *reinterpret_cast<const uint4*>(gA0 + koff);
            rA1 = *reinterpret_cast<const uint4*>(gA1 + koff);
            rB0 = *reinterpret_cast<const uint4*>(gB0 + koff);
            rB1 = *reinterpret_cast<const uint4*>(gB1 + koff);
        }
#endif
        Frag a[2], bS[2], bT[2];
#pragma unroll
        for (int i = 0; i < 2; ++i) {
            const unsigned short* p = &As[buf][(wr * 32 + i * 16 + m) * 48 + half * 8];
            a[i].u[0] = *reinterpret_cast<const uint4*>(p);
            a[i].u[1] = *reinterpret_cast<const uint4*>(p + 16);
            const unsigned short* ps = &Bt[buf][(wc * 32 + i * 16 + m) * 48 + half * 8];
            bS[i].u[0] = *reinterpret_cast<const uint4*>(ps);
            bS[i].u[1] = *reinterpret_cast<const uint4*>(ps + 16);
            const unsigned short* pt = &Bt[buf][(64 + wc * 32 + i * 16 + m) * 48 + half * 8];
            bT[i].u[0] = *reinterpret_cast<const uint4*>(pt);
            bT[i].u[1] = *reinterpret_cast<const uint4*>(pt + 16);
        }
#pragma unroll
        for (int mi = 0; mi < 2; ++mi)
#pragma unroll
            for (int ni = 0; ni < 2; ++ni) {
                accS[mi][ni] = __builtin_amdgcn_wmma_f32_16x16x32_bf16(
                    false, a[mi].v, false, bS[ni].v, (short)0, accS[mi][ni], false, false);
                accT[mi][ni] = __builtin_amdgcn_wmma_f32_16x16x32_bf16(
                    false, a[mi].v, false, bT[ni].v, (short)0, accT[mi][ni], false, false);
            }
#ifndef ASYNC_CP
        if (more) {
            *reinterpret_cast<uint4*>(&As[buf ^ 1][lA0]) = rA0;
            *reinterpret_cast<uint4*>(&As[buf ^ 1][lA1]) = rA1;
            *reinterpret_cast<uint4*>(&Bt[buf ^ 1][lA0]) = rB0;
            *reinterpret_cast<uint4*>(&Bt[buf ^ 1][lA1]) = rB1;
        }
#endif
        buf ^= 1;
    }

#pragma unroll
    for (int mi = 0; mi < 2; ++mi) {
        int rbase = row0 + wr * 32 + mi * 16 + half * 8;
#pragma unroll
        for (int ni = 0; ni < 2; ++ni) {
            int colg = col0 + wc * 32 + ni * 16 + m;           // 0..511
            float bs = b3[colg];
            float bt = b3[colg + 512];
            float ealv = 0.0f, abv = 0.0f;
            bool keep = false;
            if (EPI == 0) { ealv = __expf(alp[colg]); abv = abp[colg]; }
            else          { keep = mask_keep(colg, mt_mask); }
#pragma unroll
            for (int r = 0; r < 8; ++r) {
                float s = accS[mi][ni][r] + bs;
                float t = accT[mi][ni][r] + bt;
                size_t zi = (size_t)(rbase + r) * FDQ + colg;
                float zv = zio[zi];
                if (EPI == 0) {
                    float zc = zv * ealv + abv;
                    zio[zi] = zc * __expf(fast_tanh(s)) + t;
                } else {
                    zio[zi] = keep ? zv : (zv * __expf(fast_tanh(s)) + t);
                }
            }
        }
    }
}

// ---------------------------------------------------------------------------
// Voronoi nearest-codeword: per (b,d) scan 256 codewords of dim 16
// ---------------------------------------------------------------------------
__global__ __launch_bounds__(256)
void voronoi_kernel(const float* __restrict__ z,
                    const float* __restrict__ codebook,
                    float* __restrict__ mind,
                    int* __restrict__ idxo) {
    __shared__ float cb[NCQ * KQ];
    __shared__ float cc2[NCQ];
    const int d = blockIdx.x;
    const int tid = threadIdx.x;

    const float4* cbg = reinterpret_cast<const float4*>(codebook + ((size_t)d * NCQ + tid) * KQ);
    float ss = 0.0f;
#pragma unroll
    for (int q = 0; q < 4; ++q) {
        float4 v = cbg[q];
        cb[tid * KQ + q * 4 + 0] = v.x; cb[tid * KQ + q * 4 + 1] = v.y;
        cb[tid * KQ + q * 4 + 2] = v.z; cb[tid * KQ + q * 4 + 3] = v.w;
        ss += v.x * v.x + v.y * v.y + v.z * v.z + v.w * v.w;
    }
    cc2[tid] = ss;
    __syncthreads();

    const int b = blockIdx.y * 256 + tid;
    float p[KQ];
    float pp = 0.0f;
    const float4* zp = reinterpret_cast<const float4*>(z + (size_t)b * FDQ + d * KQ);
#pragma unroll
    for (int q = 0; q < 4; ++q) {
        float4 v = zp[q];
        p[q * 4 + 0] = v.x; p[q * 4 + 1] = v.y; p[q * 4 + 2] = v.z; p[q * 4 + 3] = v.w;
        pp += v.x * v.x + v.y * v.y + v.z * v.z + v.w * v.w;
    }
    float best = 3.402823466e38f;
    int bi = 0;
    for (int j = 0; j < NCQ; ++j) {
        float dot = 0.0f;
#pragma unroll
        for (int k = 0; k < KQ; ++k) dot += p[k] * cb[j * KQ + k];
        float d2 = pp - 2.0f * dot + cc2[j];
        if (d2 < best) { best = d2; bi = j; }
    }
    mind[(size_t)b * DQ + d] = sqrtf(fmaxf(best, 0.0f) + 1e-12f);
    idxo[(size_t)b * DQ + d] = bi;
}

// ---------------------------------------------------------------------------
// Launch
// ---------------------------------------------------------------------------
extern "C" void kernel_launch(void* const* d_in, const int* in_sizes, int n_in,
                              void* d_out, int out_size, void* d_ws, size_t ws_size,
                              hipStream_t stream) {
    (void)in_sizes; (void)n_in; (void)out_size; (void)ws_size;

    const int*   tokens  = (const int*)  d_in[0];
    const float* z_in    = (const float*)d_in[1];
    const float* emb     = (const float*)d_in[2];
    const float* codebk  = (const float*)d_in[3];
    const float* an_logs = (const float*)d_in[4];
    const float* an_b    = (const float*)d_in[5];
    const float* caW1    = (const float*)d_in[6];
    const float* caB1    = (const float*)d_in[7];
    const float* caW2    = (const float*)d_in[8];
    const float* caB2    = (const float*)d_in[9];
    const float* caW3    = (const float*)d_in[10];
    const float* caB3    = (const float*)d_in[11];
    const float* cpW1    = (const float*)d_in[12];
    const float* cpB1    = (const float*)d_in[13];
    const float* cpW2    = (const float*)d_in[14];
    const float* cpB2    = (const float*)d_in[15];
    const float* cpW3    = (const float*)d_in[16];
    const float* cpB3    = (const float*)d_in[17];

    char* ws = (char*)d_ws;
    size_t off = 0;
    auto take = [&](size_t bytes) { char* p = ws + off; off += (bytes + 255) & ~size_t(255); return p; };

    float*          z_cur = (float*)         take((size_t)BQ * FDQ * 4);
    unsigned short* cond  = (unsigned short*)take((size_t)BQ * CCQ * 2);
    unsigned short* zin   = (unsigned short*)take((size_t)BQ * CINQ * 2);
    unsigned short* h1    = (unsigned short*)take((size_t)BQ * HQ * 2);
    unsigned short* h2    = (unsigned short*)take((size_t)BQ * HQ * 2);
    unsigned short* hc1   = (unsigned short*)take((size_t)BQ * 128 * 2);
    unsigned short* hc2   = (unsigned short*)take((size_t)BQ * 128 * 2);
    unsigned short* wca1  = (unsigned short*)take((size_t)CCQ * 128 * 2);   // [128,2048]
    unsigned short* wca2  = (unsigned short*)take((size_t)128 * 128 * 2);   // [128,128]
    unsigned short* wca3  = (unsigned short*)take((size_t)128 * 1024 * 2);  // [1024,128]
    unsigned short* wcp1  = (unsigned short*)take((size_t)CINQ * HQ * 2);   // [1024,2560]
    unsigned short* wcp2  = (unsigned short*)take((size_t)HQ * HQ * 2);     // [1024,1024]
    unsigned short* wcp3  = (unsigned short*)take((size_t)HQ * 1024 * 2);   // [1024,1024]

    (void)hipMemcpyAsync(z_cur, z_in, (size_t)BQ * FDQ * 4, hipMemcpyDeviceToDevice, stream);

    {
        int n = BQ * DQ * CQ;
        gather_cond_kernel<<<n / 256, 256, 0, stream>>>(tokens, emb, cond, zin);
    }

    auto cvtT = [&](const float* s, unsigned short* d, int K, int N) {
        cvt_transpose_kernel<<<dim3(N / 32, K / 32), 256, 0, stream>>>(s, d, K, N);
    };

    for (int nb = 0; nb < NBQ; ++nb) {
        const int mt = nb & 3;
        cvtT(caW1 + (size_t)nb * CCQ * 128,  wca1, CCQ, 128);
        cvtT(caW2 + (size_t)nb * 128 * 128,  wca2, 128, 128);
        cvtT(caW3 + (size_t)nb * 128 * 1024, wca3, 128, 1024);
        cvtT(cpW1 + (size_t)nb * CINQ * HQ,  wcp1, CINQ, HQ);
        cvtT(cpW2 + (size_t)nb * HQ * HQ,    wcp2, HQ, HQ);
        cvtT(cpW3 + (size_t)nb * HQ * 1024,  wcp3, HQ, 1024);

        // conditional-affine net
        gemm_act_kernel<0><<<dim3(128 / 64, BQ / 128), 256, 0, stream>>>(
            cond, wca1, caB1 + nb * 128, hc1, CCQ, 128);
        gemm_act_kernel<0><<<dim3(128 / 64, BQ / 128), 256, 0, stream>>>(
            hc1, wca2, caB2 + nb * 128, hc2, 128, 128);
        gemm_dual_kernel<0><<<dim3(FDQ / 64, BQ / 128), 256, 0, stream>>>(
            hc2, wca3, caB3 + nb * 1024, z_cur,
            an_logs + nb * FDQ, an_b + nb * FDQ, 128, mt);

        // masked-z strips of zin
        zin_zm_kernel<<<(BQ * FDQ) / 256, 256, 0, stream>>>(z_cur, zin, mt);

        // coupling net
        gemm_act_kernel<1><<<dim3(HQ / 64, BQ / 128), 256, 0, stream>>>(
            zin, wcp1, cpB1 + nb * HQ, h1, CINQ, HQ);
        gemm_act_kernel<1><<<dim3(HQ / 64, BQ / 128), 256, 0, stream>>>(
            h1, wcp2, cpB2 + nb * HQ, h2, HQ, HQ);
        gemm_dual_kernel<1><<<dim3(FDQ / 64, BQ / 128), 256, 0, stream>>>(
            h2, wcp3, cpB3 + nb * 1024, z_cur, nullptr, nullptr, HQ, mt);
    }

    float* mind = (float*)d_out;
    int*   idxo = (int*)d_out + (size_t)BQ * DQ;
    voronoi_kernel<<<dim3(DQ, BQ / 256), 256, 0, stream>>>(z_cur, codebk, mind, idxo);
}
